// CSSMTinyViT_24412594110893
// MI455X (gfx1250) — matrix operand order repaired and checked by reading, not compiled, not run on hardware
//
#include <hip/hip_runtime.h>
#include <hip/hip_bf16.h>

// ---------------------------------------------------------------------------
// CSSM TinyViT forward for MI455X (gfx1250, wave32, WMMA bf16).
// GEMMs (≈95% of 230 GFLOP) run on v_wmma_f32_16x16x32_bf16 with f32
// accumulate. Macro-tile 128x64 (8 waves), 32x32 per wave (2x2 WMMA tiles),
// LDS ping-pong double buffering. A-tiles are staged with CDNA5 async
// global->LDS copies (global_load_async_to_lds_b128, tracked on ASYNCcnt) so
// the copy engine runs concurrently with the WMMAs; B-tiles are transposed
// through registers into LDS.
// ---------------------------------------------------------------------------

typedef __attribute__((ext_vector_type(16))) __bf16 v16bf;
typedef __attribute__((ext_vector_type(8)))  float  v8f;

#define BQ 32          // batch
#define GRID2 196      // 14*14 tokens
#define CCH 384
#define HID 1536
#define NTOK (BQ * GRID2)          // 6272 rows
#define NELT (NTOK * CCH)          // 2408448
#define KPATCH 768                 // 16*16*3

#define BM 128
#define BN 64
#define BK 32
#define LPAD 40  // padded LDS row (bf16 elems) = 80B: 16B aligned, conflict-free

enum { EPI_NONE = 0, EPI_SIGMOID = 1, EPI_GELU_BF16 = 2, EPI_ADDTO = 3 };

// ---------------------------------------------------------------------------
// Tiled bf16 WMMA GEMM:  C(MxN,f32) = A(MxK,bf16) * B(KxN,bf16) + bias
// 256 threads = 8 waves (4 M x 2 N). Each wave owns a 32x32 tile = 2x2 WMMAs.
// ---------------------------------------------------------------------------
template <int EPI>
__global__ __launch_bounds__(256) void gemm_bf16(
    const __bf16* __restrict__ A, const __bf16* __restrict__ Bw,
    const float* __restrict__ bias, float* __restrict__ Cf,
    __bf16* __restrict__ Cb, int M, int N, int K) {
  __shared__ __bf16 As[2][BM][LPAD];
  __shared__ __bf16 Bs[2][BN][LPAD];  // transposed: [n][k]

  const int tid  = threadIdx.x;
  const int lane = tid & 31;
  const int wave = tid >> 5;
  const int wm   = wave >> 1;  // 0..3
  const int wn   = wave & 1;   // 0..1
  const int m0   = blockIdx.y * BM;
  const int n0   = blockIdx.x * BN;

  // staging thread mapping
  const int arow = tid >> 1;         // 0..127
  const int acol = (tid & 1) * 16;   // 0 or 16 (bf16 elems)
  const int bkr  = tid >> 3;         // 0..31 (k within tile)
  const int bnc  = (tid & 7) * 8;    // 0..56

  // A-tile: async global->LDS copy, 32B per thread (2 x b128; the inst offset
  // applies to both the LDS and the global address).
  const __bf16* aBase = &A[(size_t)(m0 + arow) * K + acol];
  const unsigned ldsA[2] = {
      (unsigned)(size_t)(void*)&As[0][arow][acol],
      (unsigned)(size_t)(void*)&As[1][arow][acol]};
  auto asyncA = [&](int k0, int buf) {
    const unsigned long long ga = (unsigned long long)(const void*)(aBase + k0);
    asm volatile(
        "global_load_async_to_lds_b128 %0, %1, off\n\t"
        "global_load_async_to_lds_b128 %0, %1, off offset:16"
        :: "v"(ldsA[buf]), "v"(ga) : "memory");
  };
  auto waitAsync = []() {
    asm volatile("s_wait_asynccnt 0x0" ::: "memory");
  };

  // B-tile: load 8 bf16 along N, scatter-transpose into LDS.
  uint4 rb;
  auto loadB = [&](int k0) {
    rb = *(const uint4*)&Bw[(size_t)(k0 + bkr) * N + n0 + bnc];
  };
  auto storeB = [&](int buf) {
    const __bf16* pv = (const __bf16*)&rb;
#pragma unroll
    for (int i = 0; i < 8; ++i) Bs[buf][bnc + i][bkr] = pv[i];
  };

  v8f acc[2][2] = {};
  const int lr = lane & 15;
  const int lh = lane >> 4;
  const int akb = lh * 8;    // A: K = lh*8 + {0..7} and +16
  const int bkb = lh * 8;    // B: K = lh*8 + {0..7} and +16

  const int nk = K / BK;
  asyncA(0, 0);
  loadB(0);
  storeB(0);
  waitAsync();
  __syncthreads();

  for (int ks = 0; ks < nk; ++ks) {
    const int cur = ks & 1;
    const bool more = (ks + 1) < nk;
    if (more) {                      // overlap next-tile staging with compute
      asyncA((ks + 1) * BK, cur ^ 1);
      loadB((ks + 1) * BK);
    }

    union { uint4 q[2]; v16bf v; } af[2], bfr[2];
#pragma unroll
    for (int ti = 0; ti < 2; ++ti) {
      const int r = wm * 32 + ti * 16 + lr;
      af[ti].q[0] = *(const uint4*)&As[cur][r][akb];
      af[ti].q[1] = *(const uint4*)&As[cur][r][akb + 16];
    }
#pragma unroll
    for (int tj = 0; tj < 2; ++tj) {
      const int c = wn * 32 + tj * 16 + lr;
      bfr[tj].q[0] = *(const uint4*)&Bs[cur][c][bkb];
      bfr[tj].q[1] = *(const uint4*)&Bs[cur][c][bkb + 16];
    }
#pragma unroll
    for (int ti = 0; ti < 2; ++ti)
#pragma unroll
      for (int tj = 0; tj < 2; ++tj)
        acc[ti][tj] = __builtin_amdgcn_wmma_f32_16x16x32_bf16(
            false, af[ti].v, false, bfr[tj].v, (short)0, acc[ti][tj],
            false, false);

    if (more) {
      storeB(cur ^ 1);   // s_wait_loadcnt lands here, after the WMMAs
      waitAsync();       // async A copy for next tile complete
    }
    __syncthreads();
  }

  // ---- epilogue: VGPR j -> row j + 8*lh, col = lr
#pragma unroll
  for (int ti = 0; ti < 2; ++ti) {
    const int rbase = m0 + wm * 32 + ti * 16 + lh * 8;
#pragma unroll
    for (int tj = 0; tj < 2; ++tj) {
      const int col = n0 + wn * 32 + tj * 16 + lr;
      const float bv = bias[col];
#pragma unroll
      for (int j = 0; j < 8; ++j) {
        const size_t off = (size_t)(rbase + j) * N + col;
        float v = acc[ti][tj][j] + bv;
        if (EPI == EPI_NONE) {
          Cf[off] = v;
        } else if (EPI == EPI_SIGMOID) {
          Cf[off] = 1.0f / (1.0f + __expf(-v));
        } else if (EPI == EPI_GELU_BF16) {
          float t = tanhf(0.7978845608f * (v + 0.044715f * v * v * v));
          Cb[off] = (__bf16)(0.5f * v * (1.0f + t));
        } else {  // EPI_ADDTO: residual accumulate
          Cf[off] += v;
        }
      }
    }
  }
}

// ---------------------------------------------------------------------------
// LayerNorm over C=384, one wave32 per row (12 elems/lane, shfl reductions).
// MODE 0: zb = bf16(LN(h)*s+b)
// MODE 1: h += sres (residual), then zb = bf16(LN(h)*s+b)
// MODE 2: fout = LN(h)*s+b  (f32, final LN)
// ---------------------------------------------------------------------------
__device__ __forceinline__ float wave_sum(float v) {
#pragma unroll
  for (int off = 16; off > 0; off >>= 1) v += __shfl_xor(v, off, 32);
  return v;
}

template <int MODE>
__global__ __launch_bounds__(128) void ln_kernel(
    float* __restrict__ h, const float* __restrict__ sres,
    const float* __restrict__ gs, const float* __restrict__ gb,
    __bf16* __restrict__ zb, float* __restrict__ fout, int rows) {
  const int wave = threadIdx.x >> 5;
  const int lane = threadIdx.x & 31;
  const int row  = blockIdx.x * 4 + wave;
  if (row >= rows) return;
  float* hr = h + (size_t)row * CCH;
  float vals[12];
  float s1 = 0.f;
#pragma unroll
  for (int i = 0; i < 12; ++i) {
    const int c = lane + i * 32;
    float v = hr[c];
    if (MODE == 1) { v += sres[(size_t)row * CCH + c]; hr[c] = v; }
    vals[i] = v;
    s1 += v;
  }
  s1 = wave_sum(s1);
  const float mean = s1 * (1.0f / CCH);
  float s2 = 0.f;
#pragma unroll
  for (int i = 0; i < 12; ++i) { const float d = vals[i] - mean; s2 += d * d; }
  s2 = wave_sum(s2);
  const float rstd = rsqrtf(s2 * (1.0f / CCH) + 1e-6f);
#pragma unroll
  for (int i = 0; i < 12; ++i) {
    const int c = lane + i * 32;
    const float o = (vals[i] - mean) * rstd * gs[c] + gb[c];
    if (MODE == 2) fout[(size_t)row * CCH + c] = o;
    else           zb[(size_t)row * CCH + c] = (__bf16)o;
  }
}

// ---------------------------------------------------------------------------
// CSSM scan: s' = g * dwconv3x3(s) + (1-g) * u   (zero pad, per-channel)
// ---------------------------------------------------------------------------
__global__ void cssm_init(const float* __restrict__ g, const float* __restrict__ u,
                          float* __restrict__ s, int n) {
  const int i = blockIdx.x * blockDim.x + threadIdx.x;
  if (i < n) s[i] = (1.0f - g[i]) * u[i];
}

__global__ __launch_bounds__(256) void cssm_step(
    const float* __restrict__ s_in, const float* __restrict__ g,
    const float* __restrict__ u, const float* __restrict__ kd,
    float* __restrict__ s_out) {
  const int idx = blockIdx.x * blockDim.x + threadIdx.x;
  if (idx >= NELT) return;
  const int c = idx % CCH;
  int t = idx / CCH;
  const int x = t % 14; t /= 14;
  const int y = t % 14;
  const int b = t / 14;
  float a = 0.f;
#pragma unroll
  for (int dy = 0; dy < 3; ++dy) {
    const int yy = y + dy - 1;
    if (yy < 0 || yy >= 14) continue;
#pragma unroll
    for (int dx = 0; dx < 3; ++dx) {
      const int xx = x + dx - 1;
      if (xx < 0 || xx >= 14) continue;
      a += kd[(dy * 3 + dx) * CCH + c] *
           s_in[(((size_t)b * 14 + yy) * 14 + xx) * CCH + c];
    }
  }
  const float gv = g[idx];
  s_out[idx] = gv * a + (1.0f - gv) * u[idx];
}

// ---------------------------------------------------------------------------
// Misc small kernels
// ---------------------------------------------------------------------------
__global__ void f32_to_bf16(const float* __restrict__ s, __bf16* __restrict__ d,
                            size_t n) {
  const size_t i = (size_t)blockIdx.x * blockDim.x + threadIdx.x;
  if (i < n) d[i] = (__bf16)s[i];
}

__global__ void im2col_patch(const float* __restrict__ x, __bf16* __restrict__ Ap) {
  const size_t i = (size_t)blockIdx.x * blockDim.x + threadIdx.x;
  if (i >= (size_t)NTOK * KPATCH) return;
  const int k = (int)(i % KPATCH);
  const int m = (int)(i / KPATCH);
  const int b  = m / GRID2;
  const int r  = m % GRID2;
  const int gy = r / 14, gx = r % 14;
  const int py = k / 48;
  const int rem = k % 48;
  const int px = rem / 3, ch = rem % 3;
  const float v =
      x[(((size_t)b * 224 + gy * 16 + py) * 224 + gx * 16 + px) * 3 + ch];
  Ap[i] = (__bf16)v;
}

__global__ void add_pos(float* __restrict__ h, const float* __restrict__ pos, int n) {
  const int i = blockIdx.x * blockDim.x + threadIdx.x;
  if (i < n) h[i] += pos[i % (GRID2 * CCH)];
}

__global__ void maxpool_kernel(const float* __restrict__ lnh, float* __restrict__ pooled) {
  const int i = blockIdx.x * blockDim.x + threadIdx.x;
  if (i >= BQ * CCH) return;
  const int c = i % CCH, b = i / CCH;
  float m = -INFINITY;
  for (int r = 0; r < GRID2; ++r)
    m = fmaxf(m, lnh[((size_t)b * GRID2 + r) * CCH + c]);
  pooled[i] = m;
}

__global__ void head_kernel(const float* __restrict__ pooled,
                            const float* __restrict__ w,
                            const float* __restrict__ b,
                            float* __restrict__ out) {
  const int i = blockIdx.x * blockDim.x + threadIdx.x;
  if (i >= BQ * 1000) return;
  const int n = i % 1000, bb = i / 1000;
  float a = b[n];
  for (int c = 0; c < CCH; ++c) a += pooled[bb * CCH + c] * w[c * 1000 + n];
  out[i] = a;
}

// ---------------------------------------------------------------------------
// Host launcher
// ---------------------------------------------------------------------------
extern "C" void kernel_launch(void* const* d_in, const int* in_sizes, int n_in,
                              void* d_out, int out_size, void* d_ws, size_t ws_size,
                              hipStream_t stream) {
  const float* x       = (const float*)d_in[0];
  const float* patch_w = (const float*)d_in[1];
  const float* patch_b = (const float*)d_in[2];
  const float* pos     = (const float*)d_in[3];
  const float* ln1_s   = (const float*)d_in[4];
  const float* ln1_b   = (const float*)d_in[5];
  const float* w_in    = (const float*)d_in[6];
  const float* b_in    = (const float*)d_in[7];
  const float* w_g     = (const float*)d_in[8];
  const float* b_g     = (const float*)d_in[9];
  const float* k_dw    = (const float*)d_in[10];
  const float* ln2_s   = (const float*)d_in[11];
  const float* ln2_b   = (const float*)d_in[12];
  const float* w1      = (const float*)d_in[13];
  const float* b1      = (const float*)d_in[14];
  const float* w2      = (const float*)d_in[15];
  const float* b2      = (const float*)d_in[16];
  const float* lnf_s   = (const float*)d_in[17];
  const float* lnf_b   = (const float*)d_in[18];
  const float* head_w  = (const float*)d_in[19];
  const float* head_b  = (const float*)d_in[20];

  // -- workspace partition --
  char* ws = (char*)d_ws;
  auto alloc = [&](size_t bytes) -> char* {
    char* p = ws;
    ws += (bytes + 255) & ~(size_t)255;
    return p;
  };
  float*  h   = (float*)alloc((size_t)NELT * 4);
  float*  u   = (float*)alloc((size_t)NELT * 4);   // reused as final-LN buffer
  float*  g   = (float*)alloc((size_t)NELT * 4);
  float*  s0  = (float*)alloc((size_t)NELT * 4);
  float*  s1  = (float*)alloc((size_t)NELT * 4);
  __bf16* zb  = (__bf16*)alloc((size_t)NELT * 2);
  __bf16* tb  = (__bf16*)alloc((size_t)NTOK * HID * 2);
  __bf16* Ap  = (__bf16*)alloc((size_t)NTOK * KPATCH * 2);
  __bf16* Wpb = (__bf16*)alloc((size_t)KPATCH * CCH * 2);
  __bf16* Wib = (__bf16*)alloc((size_t)12 * CCH * CCH * 2);
  __bf16* Wgb = (__bf16*)alloc((size_t)12 * CCH * CCH * 2);
  __bf16* W1b = (__bf16*)alloc((size_t)12 * CCH * HID * 2);
  __bf16* W2b = (__bf16*)alloc((size_t)12 * HID * CCH * 2);
  float*  pooled = (float*)alloc((size_t)BQ * CCH * 4);

  auto blk = [](size_t n) { return (unsigned)((n + 255) / 256); };

  // -- weight conversion to bf16 --
  f32_to_bf16<<<blk((size_t)KPATCH * CCH), 256, 0, stream>>>(patch_w, Wpb, (size_t)KPATCH * CCH);
  f32_to_bf16<<<blk((size_t)12 * CCH * CCH), 256, 0, stream>>>(w_in, Wib, (size_t)12 * CCH * CCH);
  f32_to_bf16<<<blk((size_t)12 * CCH * CCH), 256, 0, stream>>>(w_g, Wgb, (size_t)12 * CCH * CCH);
  f32_to_bf16<<<blk((size_t)12 * CCH * HID), 256, 0, stream>>>(w1, W1b, (size_t)12 * CCH * HID);
  f32_to_bf16<<<blk((size_t)12 * HID * CCH), 256, 0, stream>>>(w2, W2b, (size_t)12 * HID * CCH);

  // -- patch embed: im2col + GEMM + pos --
  im2col_patch<<<blk((size_t)NTOK * KPATCH), 256, 0, stream>>>(x, Ap);
  gemm_bf16<EPI_NONE><<<dim3(CCH / BN, NTOK / BM), 256, 0, stream>>>(
      Ap, Wpb, patch_b, h, nullptr, NTOK, CCH, KPATCH);
  add_pos<<<blk(NELT), 256, 0, stream>>>(h, pos, NELT);

  // -- 12 blocks --
  for (int d = 0; d < 12; ++d) {
    ln_kernel<0><<<NTOK / 4, 128, 0, stream>>>(
        h, nullptr, ln1_s + d * CCH, ln1_b + d * CCH, zb, nullptr, NTOK);
    gemm_bf16<EPI_NONE><<<dim3(CCH / BN, NTOK / BM), 256, 0, stream>>>(
        zb, Wib + (size_t)d * CCH * CCH, b_in + d * CCH, u, nullptr, NTOK, CCH, CCH);
    gemm_bf16<EPI_SIGMOID><<<dim3(CCH / BN, NTOK / BM), 256, 0, stream>>>(
        zb, Wgb + (size_t)d * CCH * CCH, b_g + d * CCH, g, nullptr, NTOK, CCH, CCH);

    cssm_init<<<blk(NELT), 256, 0, stream>>>(g, u, s0, NELT);
    const float* kd = k_dw + (size_t)d * 9 * CCH;
    float* sa = s0;
    float* sb = s1;
    for (int t = 1; t < 8; ++t) {  // 7 remaining scan steps
      cssm_step<<<blk(NELT), 256, 0, stream>>>(sa, g, u, kd, sb);
      float* tmp = sa; sa = sb; sb = tmp;
    }

    ln_kernel<1><<<NTOK / 4, 128, 0, stream>>>(
        h, sa, ln2_s + d * CCH, ln2_b + d * CCH, zb, nullptr, NTOK);
    gemm_bf16<EPI_GELU_BF16><<<dim3(HID / BN, NTOK / BM), 256, 0, stream>>>(
        zb, W1b + (size_t)d * CCH * HID, b1 + d * HID, nullptr, tb, NTOK, HID, CCH);
    gemm_bf16<EPI_ADDTO><<<dim3(CCH / BN, NTOK / BM), 256, 0, stream>>>(
        tb, W2b + (size_t)d * HID * CCH, b2 + d * CCH, h, nullptr, NTOK, CCH, HID);
  }

  // -- final LN, max-pool, head --
  ln_kernel<2><<<NTOK / 4, 128, 0, stream>>>(h, nullptr, lnf_s, lnf_b, nullptr, u, NTOK);
  maxpool_kernel<<<blk(BQ * CCH), 256, 0, stream>>>(u, pooled);
  head_kernel<<<blk(BQ * 1000), 256, 0, stream>>>(pooled, head_w, head_b, (float*)d_out);
}
